// BAttention_45603962749199
// MI455X (gfx1250) — matrix operand order repaired and checked by reading, not compiled, not run on hardware
//
#include <hip/hip_runtime.h>
#include <hip/hip_bf16.h>
#include <stdint.h>

// ---------------------------------------------------------------------------
// BAttention for MI455X (gfx1250, wave32, WMMA 16x16x32 bf16 / f32 accum)
//
//   1) convert_x    : x -> bf16
//      transpose_w  : Wq/Wk/Wv/Wo -> transposed bf16 (LDS-tiled, coalesced)
//   2) gemm64x64 x3 : 64x64 register-blocked tiles (16 WMMA / k-step);
//                     Q (scaled 1/8) / K row-major bf16; V head-transposed
//   3) flash_attn   : one wave = (b, h, 32-query strip); online softmax over
//                     2048 kv, raw-logits f32 output as byproduct, P->LDS
//                     transpose, P@V via WMMA, K/V loads shared by 2 strips,
//                     global_prefetch of next kv chunk
//   4) gemm64x64    : attn_out @ Wo -> f32 output
//
// __launch_bounds__(128, 1) on the WMMA kernels: the 4x4 accumulator blocking
// needs ~210-240 VGPRs; without it the compiler targets default occupancy and
// spills accumulators to scratch inside the inner loop (seen in round 2).
// ---------------------------------------------------------------------------

typedef __bf16 bf16_t;
typedef __attribute__((ext_vector_type(16))) __bf16 v16bf;
typedef __attribute__((ext_vector_type(8)))  __bf16 v8bf;
typedef __attribute__((ext_vector_type(8)))  float  v8f;

#define S_LEN   2048
#define HDIM    1024
#define NHEADS  16
#define HD      64
#define BATCH   2
#define MTOT    (BATCH * S_LEN)                 // 4096 tokens
#define OUT_ELEMS ((size_t)MTOT * HDIM)         // 4,194,304

// workspace byte offsets (total 48 MiB)
#define WS_XB   ((size_t)0)          // x bf16            [4096,1024]  8 MiB
#define WS_WTQ  ((size_t)8388608)    // Wq^T bf16         [1024,1024]  2 MiB
#define WS_WTK  ((size_t)10485760)
#define WS_WTV  ((size_t)12582912)
#define WS_WTO  ((size_t)14680064)
#define WS_QB   ((size_t)16777216)   // Q bf16 (scaled)   [4096,1024]  8 MiB
#define WS_KB   ((size_t)25165824)   // K bf16            [4096,1024]  8 MiB
#define WS_VT   ((size_t)33554432)   // V^T bf16          [2,16,64,2048] 8 MiB
#define WS_AO   ((size_t)41943040)   // attn out bf16     [4096,1024]  8 MiB

static __device__ __forceinline__ v8f wmma_bf16(v16bf a, v16bf b, v8f c) {
  // (neg_a, A, neg_b, B, c_mod, C, reuse_a, reuse_b)
  return __builtin_amdgcn_wmma_f32_16x16x32_bf16(false, a, false, b, (short)0, c,
                                                 false, false);
}

// two 16-byte chunks -> one 32-byte per-lane WMMA operand
static __device__ __forceinline__ v16bf ldpair(const bf16_t* p0, const bf16_t* p1) {
  v8bf lo = *(const v8bf*)p0;
  v8bf hi = *(const v8bf*)p1;
  return __builtin_shufflevector(lo, hi, 0, 1, 2, 3, 4, 5, 6, 7,
                                         8, 9, 10, 11, 12, 13, 14, 15);
}

// deterministic counter-based hash RNG -> Weibull reparameterized noise
static __device__ __forceinline__ float weibull_noise(size_t idx) {
  uint32_t h = (uint32_t)idx * 0x9E3779B1u + (uint32_t)(idx >> 32) * 0x85EBCA77u
             + 0x6C62272Eu;
  h ^= h >> 16; h *= 0x7FEB352Du;
  h ^= h >> 15; h *= 0x846CA68Bu;
  h ^= h >> 16;
  const float u = (float)(h >> 8) * 5.9604645e-08f;        // [0,1)
  const float g = fmaf(u, 1.0f - 2.0e-5f, 1.0e-5f);        // u*(1-2eps)+eps
  return 0.1f * __logf(1.0e-5f - __logf(g));               // (1/k)*log(-log(g)+eps)
}

// ---------------------------------------------------------------------------
__global__ void convert_x(const float* __restrict__ x, bf16_t* __restrict__ xb) {
  const size_t i = (size_t)blockIdx.x * blockDim.x + threadIdx.x;
  xb[i] = (bf16_t)x[i];
}

// LDS-tiled transpose: wt[n][k] = (bf16) w[k][n]; coalesced both sides.
__global__ void transpose_w(const float* __restrict__ wq, const float* __restrict__ wk,
                            const float* __restrict__ wv, const float* __restrict__ wo,
                            bf16_t* __restrict__ wtq, bf16_t* __restrict__ wtk,
                            bf16_t* __restrict__ wtv, bf16_t* __restrict__ wto) {
  __shared__ float tile[32][33];                // pitch 33: bank-conflict free
  const int which = blockIdx.y;
  const float* w = (which == 0) ? wq : (which == 1) ? wk : (which == 2) ? wv : wo;
  bf16_t* wt = (which == 0) ? wtq : (which == 1) ? wtk : (which == 2) ? wtv : wto;
  const int tx = threadIdx.x, ty = threadIdx.y;        // 32 x 8
  const int kb0 = (blockIdx.x >> 5) * 32;              // k tile
  const int nb0 = (blockIdx.x & 31) * 32;              // n tile
#pragma unroll
  for (int r = 0; r < 4; ++r)
    tile[ty + r * 8][tx] = w[(size_t)(kb0 + ty + r * 8) * HDIM + nb0 + tx];
  __syncthreads();
#pragma unroll
  for (int r = 0; r < 4; ++r)
    wt[(size_t)(nb0 + ty + r * 8) * HDIM + kb0 + tx] = (bf16_t)tile[tx][ty + r * 8];
}

// ---------------------------------------------------------------------------
// C[m,n] = sum_k A[m,k]*Bt[n,k]; one wave owns a 64x64 tile (4x4 WMMA tiles).
// mode 0: bf16 row-major store (scale applied)   -> Q (0.125) / K (1.0)
// mode 1: bf16 head-transposed store [b,h,d,s]   -> V
// mode 2: f32 row-major store                    -> final output
__global__ void __launch_bounds__(128, 1)
gemm64x64(const bf16_t* __restrict__ A, const bf16_t* __restrict__ Bt,
          bf16_t* __restrict__ outb, float* __restrict__ outf,
          int mode, float scale) {
  const int lane = threadIdx.x & 31;
  const int l15 = lane & 15, half = lane >> 4;
  const int wid = blockIdx.x * (blockDim.x >> 5) + (threadIdx.x >> 5);   // 0..1023
  const int mbase = (wid >> 4) * 64;       // 64 m-supertiles
  const int nbase = (wid & 15) * 64;       // 16 n-supertiles

  const bf16_t* ar = A  + (size_t)(mbase + l15) * HDIM + half * 8;
  const bf16_t* br = Bt + (size_t)(nbase + l15) * HDIM + half * 8;

  v8f c[4][4];
  const v8f vzero = {};
#pragma unroll
  for (int i = 0; i < 4; ++i)
#pragma unroll
    for (int j = 0; j < 4; ++j) c[i][j] = vzero;

  for (int kc = 0; kc < HDIM / 32; ++kc) {
    v16bf av[4], bv[4];
#pragma unroll
    for (int i = 0; i < 4; ++i) {
      const bf16_t* p = ar + (size_t)i * 16 * HDIM;
      av[i] = ldpair(p, p + 16);           // A: K {0..7}/{16..23} per half
    }
#pragma unroll
    for (int j = 0; j < 4; ++j) {
      const bf16_t* p = br + (size_t)j * 16 * HDIM;
      bv[j] = ldpair(p, p + 16);           // B: K {0..7}/{16..23} per half
    }
#pragma unroll
    for (int i = 0; i < 4; ++i)
#pragma unroll
      for (int j = 0; j < 4; ++j) c[i][j] = wmma_bf16(av[i], bv[j], c[i][j]);
    ar += 32; br += 32;
  }

#pragma unroll
  for (int i = 0; i < 4; ++i) {
    const int mrow = mbase + i * 16 + half * 8;      // C: VGPR r -> row mrow+r
#pragma unroll
    for (int j = 0; j < 4; ++j) {
      const int ncol = nbase + j * 16 + l15;
      if (mode == 0) {
#pragma unroll
        for (int r = 0; r < 8; ++r)
          outb[(size_t)(mrow + r) * HDIM + ncol] = (bf16_t)(c[i][j][r] * scale);
      } else if (mode == 1) {
        const int h = ncol >> 6, d = ncol & 63;
        const int bb = mrow >> 11, s = mrow & 2047;  // tile never crosses batch
        v8bf pk;
#pragma unroll
        for (int r = 0; r < 8; ++r) pk[r] = (bf16_t)c[i][j][r];
        *(v8bf*)(outb + ((size_t)(bb * NHEADS + h) * HD + d) * S_LEN + s) = pk;
      } else {
#pragma unroll
        for (int r = 0; r < 8; ++r)
          outf[(size_t)(mrow + r) * HDIM + ncol] = c[i][j][r];
      }
    }
  }
}

// ---------------------------------------------------------------------------
// One wave per (b, h, 32-query strip): two 16-row strips share K/V loads.
__global__ void __launch_bounds__(128, 1)
flash_attn(const bf16_t* __restrict__ qb, const bf16_t* __restrict__ kb,
           const bf16_t* __restrict__ vt, const float* __restrict__ qmask,
           bf16_t* __restrict__ aob, float* __restrict__ logits) {
  __shared__ bf16_t lds[4][32][40];      // per-wave P tiles, pitch 40 (bank-safe)
  const int lane = threadIdx.x & 31;
  const int l15 = lane & 15, half = lane >> 4;
  const int wslot = threadIdx.x >> 5;
  const int wid = blockIdx.x * 4 + wslot;       // 0..2047
  const int qt = wid & 63;                      // S/32 strips
  const int bh = wid >> 6;
  const int h = bh & 15, b = bh >> 4;
  const int qbase = qt * 32;

  // Q strips as A operands (hd 0..31, 32..63), resident all loop
  v16bf aq[2][2];
  float qmq[2][8], m8[2][8], l8[2][8];
  v8f acc[2][4];
  const v8f vzero = {};
#pragma unroll
  for (int s = 0; s < 2; ++s) {
    const bf16_t* qrow =
        qb + (size_t)(b * S_LEN + qbase + s * 16 + l15) * HDIM + h * HD + half * 8;
    aq[s][0] = ldpair(qrow, qrow + 16);
    aq[s][1] = ldpair(qrow + 32, qrow + 48);
#pragma unroll
    for (int r = 0; r < 8; ++r) {
      qmq[s][r] = qmask[b * S_LEN + qbase + s * 16 + half * 8 + r];
      m8[s][r] = -3.0e38f;
      l8[s][r] = 0.0f;
    }
#pragma unroll
    for (int t = 0; t < 4; ++t) acc[s][t] = vzero;
  }

  const float GL = -0.04987244125983974f;       // lgamma(1 + 1/10)
  const size_t lrow0 = (size_t)(b * NHEADS + h) * S_LEN + qbase;

  for (int c = 0; c < S_LEN / 32; ++c) {
    const int kv0 = c * 32;
    // K operands for this 32-kv chunk (shared by both q strips)
    v16bf kB[2][2];
    float qmk[2];
#pragma unroll
    for (int t = 0; t < 2; ++t) {
      const bf16_t* krow =
          kb + (size_t)(b * S_LEN + kv0 + t * 16 + l15) * HDIM + h * HD + half * 8;
      kB[t][0] = ldpair(krow, krow + 16);
      kB[t][1] = ldpair(krow + 32, krow + 48);
      qmk[t] = qmask[b * S_LEN + kv0 + t * 16 + l15];
    }
    if (c + 1 < S_LEN / 32) {   // cover next chunk's latency (global_prefetch_b8)
      __builtin_prefetch(kb + (size_t)(b * S_LEN + kv0 + 32 + lane) * HDIM + h * HD, 0, 1);
      __builtin_prefetch(vt + ((size_t)(b * NHEADS + h) * HD + lane) * S_LEN + kv0 + 32, 0, 1);
      __builtin_prefetch(vt + ((size_t)(b * NHEADS + h) * HD + 32 + lane) * S_LEN + kv0 + 32, 0, 1);
    }
#pragma unroll
    for (int s = 0; s < 2; ++s) {
      float p0[8], p1[8];
#pragma unroll
      for (int t = 0; t < 2; ++t) {
        v8f cc = vzero;
        cc = wmma_bf16(aq[s][0], kB[t][0], cc);
        cc = wmma_bf16(aq[s][1], kB[t][1], cc);
        float* pp = t ? p1 : p0;
#pragma unroll
        for (int r = 0; r < 8; ++r) {
          const float lv = cc[r];
          const size_t li =
              (lrow0 + s * 16 + half * 8 + r) * S_LEN + kv0 + t * 16 + l15;
          logits[li] = lv;                     // raw logits output (coalesced)
          pp[r] = lv - 100000.0f * (1.0f - qmq[s][r] * qmk[t]) - GL + weibull_noise(li);
        }
      }
      // chunk row-max across the 16 lanes of this half
      float tm[8];
#pragma unroll
      for (int r = 0; r < 8; ++r) tm[r] = fmaxf(p0[r], p1[r]);
#pragma unroll
      for (int m = 1; m < 16; m <<= 1) {
#pragma unroll
        for (int r = 0; r < 8; ++r) tm[r] = fmaxf(tm[r], __shfl_xor(tm[r], m, 32));
      }
      float sc[8], rs[8];
#pragma unroll
      for (int r = 0; r < 8; ++r) {
        const float nm = fmaxf(m8[s][r], tm[r]);
        sc[r] = __expf(m8[s][r] - nm);
        m8[s][r] = nm;
      }
#pragma unroll
      for (int r = 0; r < 8; ++r) {
        p0[r] = __expf(p0[r] - m8[s][r]);
        p1[r] = __expf(p1[r] - m8[s][r]);
        rs[r] = p0[r] + p1[r];
      }
#pragma unroll
      for (int m = 1; m < 16; m <<= 1) {
#pragma unroll
        for (int r = 0; r < 8; ++r) rs[r] += __shfl_xor(rs[r], m, 32);
      }
#pragma unroll
      for (int r = 0; r < 8; ++r) l8[s][r] = l8[s][r] * sc[r] + rs[r];
#pragma unroll
      for (int t = 0; t < 4; ++t)
#pragma unroll
        for (int r = 0; r < 8; ++r) acc[s][t][r] *= sc[r];
      // P tile: C-layout -> row-major LDS (transpose for the next A operand)
#pragma unroll
      for (int r = 0; r < 8; ++r) {
        lds[wslot][s * 16 + half * 8 + r][l15]      = (bf16_t)p0[r];
        lds[wslot][s * 16 + half * 8 + r][16 + l15] = (bf16_t)p1[r];
      }
    }
    __syncthreads();   // uniform trip count; orders ds stores vs ds loads
    // V operands loaded once, used by both strips
    v16bf vB[4];
#pragma unroll
    for (int t = 0; t < 4; ++t) {
      const bf16_t* vrow =
          vt + ((size_t)(b * NHEADS + h) * HD + t * 16 + l15) * S_LEN + kv0 + half * 8;
      vB[t] = ldpair(vrow, vrow + 16);
    }
#pragma unroll
    for (int s = 0; s < 2; ++s) {
      const bf16_t* prow = &lds[wslot][s * 16 + l15][half * 8];
      const v16bf pa = ldpair(prow, prow + 16);
#pragma unroll
      for (int t = 0; t < 4; ++t) acc[s][t] = wmma_bf16(pa, vB[t], acc[s][t]);
    }
    __syncthreads();
  }
  // epilogue: normalize, store attn-out bf16 [token, h*64+d]
#pragma unroll
  for (int s = 0; s < 2; ++s)
#pragma unroll
    for (int r = 0; r < 8; ++r) {
      const float inv = 1.0f / l8[s][r];
      const size_t row =
          (size_t)(b * S_LEN + qbase + s * 16 + half * 8 + r) * HDIM + h * HD;
#pragma unroll
      for (int t = 0; t < 4; ++t)
        aob[row + t * 16 + l15] = (bf16_t)(acc[s][t][r] * inv);
    }
}

// ---------------------------------------------------------------------------
extern "C" void kernel_launch(void* const* d_in, const int* in_sizes, int n_in,
                              void* d_out, int out_size, void* d_ws, size_t ws_size,
                              hipStream_t stream) {
  (void)in_sizes; (void)n_in; (void)out_size; (void)ws_size;
  const float* x  = (const float*)d_in[0];
  const float* qm = (const float*)d_in[1];
  const float* wq = (const float*)d_in[2];
  const float* wk = (const float*)d_in[3];
  const float* wv = (const float*)d_in[4];
  const float* wo = (const float*)d_in[5];

  char* ws = (char*)d_ws;
  bf16_t* xb   = (bf16_t*)(ws + WS_XB);
  bf16_t* wtq  = (bf16_t*)(ws + WS_WTQ);
  bf16_t* wtk  = (bf16_t*)(ws + WS_WTK);
  bf16_t* wtv  = (bf16_t*)(ws + WS_WTV);
  bf16_t* wto  = (bf16_t*)(ws + WS_WTO);
  bf16_t* qbuf = (bf16_t*)(ws + WS_QB);
  bf16_t* kbuf = (bf16_t*)(ws + WS_KB);
  bf16_t* vtb  = (bf16_t*)(ws + WS_VT);
  bf16_t* aob  = (bf16_t*)(ws + WS_AO);

  float* out_f  = (float*)d_out;
  float* logits = out_f + OUT_ELEMS;

  // 1) precision conversion + coalesced tiled weight transposes
  convert_x<<<16384, 256, 0, stream>>>(x, xb);
  transpose_w<<<dim3(1024, 4), dim3(32, 8), 0, stream>>>(wq, wk, wv, wo,
                                                         wtq, wtk, wtv, wto);
  // 2) QKV projections (1024 tile-waves each; 1/sqrt(64) folded into Q)
  gemm64x64<<<256, 128, 0, stream>>>(xb, wtq, qbuf, nullptr, 0, 0.125f);
  gemm64x64<<<256, 128, 0, stream>>>(xb, wtk, kbuf, nullptr, 0, 1.0f);
  gemm64x64<<<256, 128, 0, stream>>>(xb, wtv, vtb,  nullptr, 1, 1.0f);
  // 3) fused logits + noisy softmax + P@V (2048 strip-waves)
  flash_attn<<<512, 128, 0, stream>>>(qbuf, kbuf, vtb, qm, aob, logits);
  // 4) output projection
  gemm64x64<<<256, 128, 0, stream>>>(aob, wto, nullptr, out_f, 2, 1.0f);
}
